// ContextAwareAttention_35966056137119
// MI455X (gfx1250) — compile-verified, matrix operand order checked by reading, not run on hardware
//
#include <hip/hip_runtime.h>
#include <math.h>

// ---------- types ----------
typedef __attribute__((ext_vector_type(16))) __bf16 bf16x16;
typedef __attribute__((ext_vector_type(8)))  float  f32x8;
typedef __attribute__((ext_vector_type(4)))  unsigned int u32x4;

union BFrag { unsigned short u[16]; u32x4 q[2]; bf16x16 v; };

__device__ __forceinline__ unsigned short tobf16(float f) {
  unsigned int u = __float_as_uint(f);
  unsigned int r = u + 0x7fffu + ((u >> 16) & 1u);
  return (unsigned short)(r >> 16);
}

__device__ __forceinline__ float blk_sum(float v, float* sred) {
  int t = threadIdx.x;
  sred[t] = v; __syncthreads();
  for (int s = blockDim.x >> 1; s > 0; s >>= 1) {
    if (t < s) sred[t] += sred[t + s];
    __syncthreads();
  }
  float r = sred[0]; __syncthreads();
  return r;
}

// ---------- embedding + LayerNorm ----------
__global__ __launch_bounds__(256) void embed_ln_kernel(
    const int* __restrict__ ids, const float* __restrict__ emb,
    const float* __restrict__ pos, const float* __restrict__ g,
    const float* __restrict__ be, float* __restrict__ x,
    unsigned short* __restrict__ xb) {
  __shared__ float sred[256];
  int row = blockIdx.x;
  int s = row & 2047;
  int id = ids[row];
  float v0[3];
  float sum = 0.f;
  for (int i = 0; i < 3; ++i) {
    int d = threadIdx.x + i * 256;
    float val = emb[(size_t)id * 768 + d] + pos[(size_t)s * 768 + d];
    v0[i] = val; sum += val;
  }
  sum = blk_sum(sum, sred);
  float mean = sum * (1.f / 768.f);
  float sq = 0.f;
  for (int i = 0; i < 3; ++i) { float dd = v0[i] - mean; sq += dd * dd; }
  sq = blk_sum(sq, sred);
  float inv = rsqrtf(sq * (1.f / 768.f) + 1e-12f);
  for (int i = 0; i < 3; ++i) {
    int d = threadIdx.x + i * 256;
    float y = (v0[i] - mean) * inv * g[d] + be[d];
    x[(size_t)row * 768 + d] = y;
    xb[(size_t)row * 768 + d] = tobf16(y);
  }
}

// ---------- residual add + LayerNorm (in place on x/xb) ----------
__global__ __launch_bounds__(256) void add_ln_kernel(
    float* __restrict__ x, const float* __restrict__ t,
    const float* __restrict__ g, const float* __restrict__ be,
    unsigned short* __restrict__ xb) {
  __shared__ float sred[256];
  int row = blockIdx.x;
  float v0[3];
  float sum = 0.f;
  for (int i = 0; i < 3; ++i) {
    int d = threadIdx.x + i * 256;
    float val = x[(size_t)row * 768 + d] + t[(size_t)row * 768 + d];
    v0[i] = val; sum += val;
  }
  sum = blk_sum(sum, sred);
  float mean = sum * (1.f / 768.f);
  float sq = 0.f;
  for (int i = 0; i < 3; ++i) { float dd = v0[i] - mean; sq += dd * dd; }
  sq = blk_sum(sq, sred);
  float inv = rsqrtf(sq * (1.f / 768.f) + 1e-12f);
  for (int i = 0; i < 3; ++i) {
    int d = threadIdx.x + i * 256;
    float y = (v0[i] - mean) * inv * g[d] + be[d];
    x[(size_t)row * 768 + d] = y;
    xb[(size_t)row * 768 + d] = tobf16(y);
  }
}

// ---------- f32 -> bf16 conversion ----------
__global__ void f2bf_kernel(const float* __restrict__ src,
                            unsigned short* __restrict__ dst, int n) {
  int i = blockIdx.x * blockDim.x + threadIdx.x;
  if (i < n) dst[i] = tobf16(src[i]);
}

// src[R][C] -> dst[C][R], bf16 out (builds W^T so GEMM B-tiles are K-contiguous)
__global__ void transpose_f2bf_kernel(const float* __restrict__ src,
                                      unsigned short* __restrict__ dst,
                                      int R, int C) {
  int i = blockIdx.x * blockDim.x + threadIdx.x;
  if (i >= R * C) return;
  int r = i / C, c = i % C;
  dst[(size_t)c * R + r] = tobf16(src[i]);
}

// vb [8192][768] -> vT [(b,h,dh)][2048] bf16, so attention V fragments are K-contiguous
__global__ void vtrans_kernel(const unsigned short* __restrict__ vbsrc,
                              unsigned short* __restrict__ vT) {
  int idx = blockIdx.x * blockDim.x + threadIdx.x;
  if (idx >= 8192 * 768) return;
  int row = idx / 768;  // b*2048 + s
  int col = idx % 768;  // h*64 + dh
  int b = row >> 11, s = row & 2047;
  int h = col >> 6, dh = col & 63;
  vT[((size_t)((b * 12 + h) * 64 + dh) * 2048) + s] = vbsrc[idx];
}

// ---------- bf16 WMMA GEMM: C[M,N] = act(A[M,K] @ W[K,N] + bias (+bias2)) ----------
// A bf16 row-major [M][K]; WT bf16 = W transposed, row-major [N][K]. N%128==0, K%32==0.
// block = 256 thr = 8 waves (4 m-waves x 2 n-waves); wave tile 32x64; block tile 128x128.
// Double-buffered LDS tiles filled with CDNA5 async global->LDS b128 (ASYNCcnt):
// issue tile i+1 into buffer p^1 right after the barrier, compute tile i from buffer p.
template <int ACT, bool BIAS2, bool WF, bool WB>
__global__ __launch_bounds__(256) void gemm_bf16_kernel(
    const unsigned short* __restrict__ A, const unsigned short* __restrict__ WT,
    const float* __restrict__ bias, const float* __restrict__ bias2,
    float* __restrict__ Cf, unsigned short* __restrict__ Cb,
    int M, int N, int K) {
  __shared__ __align__(16) unsigned short As[2][128][40];  // [buf][m][k], 80B rows
  __shared__ __align__(16) unsigned short Ws[2][128][40];  // [buf][n][k], 80B rows
  int tid = threadIdx.x;
  int wid = tid >> 5, lane = tid & 31;
  int ln16 = lane & 15, lh = lane >> 4;
  int wm = wid >> 1, wn = wid & 1;
  int row0 = blockIdx.y * 128, col0 = blockIdx.x * 128;

  f32x8 acc[2][4];
  for (int mt = 0; mt < 2; ++mt)
    for (int nt = 0; nt < 4; ++nt)
      acc[mt][nt] = f32x8{0.f, 0.f, 0.f, 0.f, 0.f, 0.f, 0.f, 0.f};

  int arow = tid >> 1, aseg = (tid & 1) * 16;           // 2 threads per 32-k row
  int gra = row0 + arow; if (gra >= M) gra = M - 1;     // clamped; OOB C rows store-guarded
  const unsigned short* aptr = A + (size_t)gra * K + aseg;
  const unsigned short* wptr = WT + (size_t)(col0 + arow) * K + aseg;

#if defined(__gfx1250__)
  auto issue = [&](int k0, int p) {
    unsigned lda = (unsigned)(size_t)(const void*)&As[p][arow][aseg];
    unsigned long long ga = (unsigned long long)(size_t)(aptr + k0);
    asm volatile("global_load_async_to_lds_b128 %0, %1, off"
                 :: "v"(lda), "v"(ga) : "memory");
    asm volatile("global_load_async_to_lds_b128 %0, %1, off offset:16"
                 :: "v"(lda), "v"(ga) : "memory");
    unsigned ldw = (unsigned)(size_t)(const void*)&Ws[p][arow][aseg];
    unsigned long long gw = (unsigned long long)(size_t)(wptr + k0);
    asm volatile("global_load_async_to_lds_b128 %0, %1, off"
                 :: "v"(ldw), "v"(gw) : "memory");
    asm volatile("global_load_async_to_lds_b128 %0, %1, off offset:16"
                 :: "v"(ldw), "v"(gw) : "memory");
    if (k0 + 32 < K) __builtin_prefetch(aptr + k0 + 32, 0, 1);
  };
  issue(0, 0);
  int p = 0;
  for (int k0 = 0; k0 < K; k0 += 32) {
    asm volatile("s_wait_asynccnt 0" ::: "memory");
    __syncthreads();
    if (k0 + 32 < K) issue(k0 + 32, p ^ 1);
#else
  int p = 0;
  for (int k0 = 0; k0 < K; k0 += 32) {
    const u32x4* pa = reinterpret_cast<const u32x4*>(aptr + k0);
    *reinterpret_cast<u32x4*>(&As[p][arow][aseg]) = pa[0];
    *reinterpret_cast<u32x4*>(&As[p][arow][aseg + 8]) = pa[1];
    const u32x4* pw = reinterpret_cast<const u32x4*>(wptr + k0);
    *reinterpret_cast<u32x4*>(&Ws[p][arow][aseg]) = pw[0];
    *reinterpret_cast<u32x4*>(&Ws[p][arow][aseg + 8]) = pw[1];
    __syncthreads();
#endif
    // fragments: 2x b128 LDS loads each (contiguous per-lane k-runs)
    BFrag a0, a1, bb;
    {
      const unsigned short* p0 = &As[p][wm * 32 + ln16][0];
      a0.q[0] = *reinterpret_cast<const u32x4*>(p0 + lh * 8);
      a0.q[1] = *reinterpret_cast<const u32x4*>(p0 + 16 + lh * 8);
      const unsigned short* p1 = &As[p][wm * 32 + 16 + ln16][0];
      a1.q[0] = *reinterpret_cast<const u32x4*>(p1 + lh * 8);
      a1.q[1] = *reinterpret_cast<const u32x4*>(p1 + 16 + lh * 8);
    }
    for (int nt = 0; nt < 4; ++nt) {
      const unsigned short* pb = &Ws[p][wn * 64 + nt * 16 + ln16][0];
      bb.q[0] = *reinterpret_cast<const u32x4*>(pb + lh * 8);
      bb.q[1] = *reinterpret_cast<const u32x4*>(pb + 16 + lh * 8);
      acc[0][nt] = __builtin_amdgcn_wmma_f32_16x16x32_bf16(
          false, a0.v, false, bb.v, (short)0, acc[0][nt], false, false);
      acc[1][nt] = __builtin_amdgcn_wmma_f32_16x16x32_bf16(
          false, a1.v, false, bb.v, (short)0, acc[1][nt], false, false);
    }
    p ^= 1;
#if !defined(__gfx1250__)
    __syncthreads();
#endif
  }

  for (int mt = 0; mt < 2; ++mt)
    for (int nt = 0; nt < 4; ++nt)
      for (int r = 0; r < 8; ++r) {
        int m = row0 + wm * 32 + mt * 16 + r + 8 * lh;
        if (m >= M) continue;
        int n = col0 + wn * 64 + nt * 16 + ln16;
        float vl = acc[mt][nt][r] + bias[n];
        if (BIAS2) vl += bias2[n];
        if (ACT == 1) vl = 0.5f * vl * (1.f + erff(vl * 0.70710678118f));
        if (WF) Cf[(size_t)m * N + n] = vl;
        if (WB) Cb[(size_t)m * N + n] = tobf16(vl);
      }
}

// ---------- BigBird block-sparse attention ----------
// grid = (32 blocks, 12 heads, 4 batch); block = 128 thr = 4 waves, 2 key blocks/wave.
// Two-pass recompute softmax; all Q/K^T/V fragments are wide b128 loads.
__global__ __launch_bounds__(128) void attn_kernel(
    const unsigned short* __restrict__ q, const unsigned short* __restrict__ k,
    const unsigned short* __restrict__ vT, const int* __restrict__ rnd,
    unsigned short* __restrict__ o) {
  const int D = 768, S = 2048, NB = 32;
  int i = blockIdx.x, h = blockIdx.y, b = blockIdx.z;
  int tid = threadIdx.x;
  int wid = tid >> 5, lane = tid & 31;
  int ln16 = lane & 15, lh = lane >> 4;

  __shared__ __align__(16) unsigned short wbuf[4][64][72];  // per-wave prob tile, 144B rows
  __shared__ float osum[64][64];
  __shared__ float red[4][64];
  __shared__ float gval[64];
  __shared__ int gidx[8];

  for (int t = tid; t < 64 * 64; t += 128) (&osum[0][0])[t] = 0.f;
  if (tid < 8) {
    int gi;
    if (tid == 0) gi = 0;
    else if (tid == 1) gi = NB - 1;
    else if (tid == 2) gi = (i - 1 + NB) % NB;
    else if (tid == 3) gi = i;
    else if (tid == 4) gi = (i + 1) % NB;
    else gi = rnd[i * 3 + (tid - 5)];
    gidx[tid] = gi;
  }
  __syncthreads();

  const unsigned short* qbase = q + ((size_t)(b * S + i * 64)) * D + h * 64;

  BFrag qa[4][2];
  for (int mt = 0; mt < 4; ++mt)
    for (int ks = 0; ks < 2; ++ks) {
      const unsigned short* p = qbase + (size_t)(mt * 16 + ln16) * D + ks * 32;
      qa[mt][ks].q[0] = *reinterpret_cast<const u32x4*>(p + lh * 8);
      qa[mt][ks].q[1] = *reinterpret_cast<const u32x4*>(p + 16 + lh * 8);
    }

  // ---- pass 1: row max over all 8 gathered blocks ----
  float rm[4][8];
  for (int mt = 0; mt < 4; ++mt)
    for (int r = 0; r < 8; ++r) rm[mt][r] = -3.4e38f;
  for (int g2 = 0; g2 < 2; ++g2) {
    int blk = gidx[wid * 2 + g2];
    const unsigned short* kbase = k + ((size_t)(b * S + blk * 64)) * D + h * 64;
    for (int nt = 0; nt < 4; ++nt) {
      BFrag kb0, kb1;
      const unsigned short* p = kbase + (size_t)(nt * 16 + ln16) * D;  // K^T by addressing
      kb0.q[0] = *reinterpret_cast<const u32x4*>(p + lh * 8);
      kb0.q[1] = *reinterpret_cast<const u32x4*>(p + 16 + lh * 8);
      kb1.q[0] = *reinterpret_cast<const u32x4*>(p + 32 + lh * 8);
      kb1.q[1] = *reinterpret_cast<const u32x4*>(p + 48 + lh * 8);
      for (int mt = 0; mt < 4; ++mt) {
        f32x8 acc = f32x8{0.f, 0.f, 0.f, 0.f, 0.f, 0.f, 0.f, 0.f};
        acc = __builtin_amdgcn_wmma_f32_16x16x32_bf16(
            false, qa[mt][0].v, false, kb0.v, (short)0, acc, false, false);
        acc = __builtin_amdgcn_wmma_f32_16x16x32_bf16(
            false, qa[mt][1].v, false, kb1.v, (short)0, acc, false, false);
        for (int r = 0; r < 8; ++r) rm[mt][r] = fmaxf(rm[mt][r], acc[r] * 0.125f);
      }
    }
  }
  for (int mt = 0; mt < 4; ++mt)
    for (int r = 0; r < 8; ++r) {
      float m = rm[mt][r];
      for (int ml = 8; ml >= 1; ml >>= 1) m = fmaxf(m, __shfl_xor(m, ml, 16));
      if (ln16 == 0) red[wid][mt * 16 + r + 8 * lh] = m;
    }
  __syncthreads();
  if (tid < 64)
    gval[tid] = fmaxf(fmaxf(red[0][tid], red[1][tid]), fmaxf(red[2][tid], red[3][tid]));
  __syncthreads();

  // ---- pass 2: recompute, exp, P@V with unnormalized weights ----
  float rs[4][8];
  for (int mt = 0; mt < 4; ++mt)
    for (int r = 0; r < 8; ++r) rs[mt][r] = 0.f;
  for (int g2 = 0; g2 < 2; ++g2) {
    int blk = gidx[wid * 2 + g2];
    const unsigned short* kbase = k + ((size_t)(b * S + blk * 64)) * D + h * 64;
    f32x8 sc[16];
    for (int nt = 0; nt < 4; ++nt) {
      BFrag kb0, kb1;
      const unsigned short* p = kbase + (size_t)(nt * 16 + ln16) * D;
      kb0.q[0] = *reinterpret_cast<const u32x4*>(p + lh * 8);
      kb0.q[1] = *reinterpret_cast<const u32x4*>(p + 16 + lh * 8);
      kb1.q[0] = *reinterpret_cast<const u32x4*>(p + 32 + lh * 8);
      kb1.q[1] = *reinterpret_cast<const u32x4*>(p + 48 + lh * 8);
      for (int mt = 0; mt < 4; ++mt) {
        f32x8 acc = f32x8{0.f, 0.f, 0.f, 0.f, 0.f, 0.f, 0.f, 0.f};
        acc = __builtin_amdgcn_wmma_f32_16x16x32_bf16(
            false, qa[mt][0].v, false, kb0.v, (short)0, acc, false, false);
        acc = __builtin_amdgcn_wmma_f32_16x16x32_bf16(
            false, qa[mt][1].v, false, kb1.v, (short)0, acc, false, false);
        for (int r = 0; r < 8; ++r) sc[mt * 4 + nt][r] = acc[r] * 0.125f;
      }
    }
    for (int mt = 0; mt < 4; ++mt)
      for (int nt = 0; nt < 4; ++nt)
        for (int r = 0; r < 8; ++r) {
          int row = mt * 16 + r + 8 * lh;
          float e = __expf(sc[mt * 4 + nt][r] - gval[row]);
          rs[mt][r] += e;
          wbuf[wid][row][nt * 16 + ln16] = tobf16(e);
        }
    __syncthreads();
    BFrag wa[4][2];
    for (int mt = 0; mt < 4; ++mt)
      for (int ks = 0; ks < 2; ++ks) {
        const unsigned short* p = &wbuf[wid][mt * 16 + ln16][ks * 32];
        wa[mt][ks].q[0] = *reinterpret_cast<const u32x4*>(p + lh * 8);
        wa[mt][ks].q[1] = *reinterpret_cast<const u32x4*>(p + 16 + lh * 8);
      }
    const unsigned short* vtb =
        vT + ((size_t)(b * 12 + h) * 64) * 2048 + blk * 64;
    for (int nt = 0; nt < 4; ++nt) {
      BFrag vb0, vb1;
      const unsigned short* p0 = vtb + (size_t)(nt * 16 + ln16) * 2048;
      vb0.q[0] = *reinterpret_cast<const u32x4*>(p0 + lh * 8);
      vb0.q[1] = *reinterpret_cast<const u32x4*>(p0 + 16 + lh * 8);
      vb1.q[0] = *reinterpret_cast<const u32x4*>(p0 + 32 + lh * 8);
      vb1.q[1] = *reinterpret_cast<const u32x4*>(p0 + 48 + lh * 8);
      for (int mt = 0; mt < 4; ++mt) {
        f32x8 acc = f32x8{0.f, 0.f, 0.f, 0.f, 0.f, 0.f, 0.f, 0.f};
        acc = __builtin_amdgcn_wmma_f32_16x16x32_bf16(
            false, wa[mt][0].v, false, vb0.v, (short)0, acc, false, false);
        acc = __builtin_amdgcn_wmma_f32_16x16x32_bf16(
            false, wa[mt][1].v, false, vb1.v, (short)0, acc, false, false);
        for (int r = 0; r < 8; ++r)
          atomicAdd(&osum[mt * 16 + r + 8 * lh][nt * 16 + ln16], acc[r]);
      }
    }
    __syncthreads();
  }

  // cross-wave row sums, then normalized output
  for (int mt = 0; mt < 4; ++mt)
    for (int r = 0; r < 8; ++r) {
      float s = rs[mt][r];
      for (int ml = 8; ml >= 1; ml >>= 1) s += __shfl_xor(s, ml, 16);
      if (ln16 == 0) red[wid][mt * 16 + r + 8 * lh] = s;
    }
  __syncthreads();
  if (tid < 64) gval[tid] = red[0][tid] + red[1][tid] + red[2][tid] + red[3][tid];
  __syncthreads();
  for (int t = tid; t < 64 * 64; t += 128) {
    int m = t >> 6, n = t & 63;
    o[((size_t)(b * S + i * 64 + m)) * D + h * 64 + n] = tobf16(osum[m][n] / gval[m]);
  }
}

// ---------- conv1d (VALID) ----------
__global__ void conv_kernel(const float* __restrict__ x, const float* __restrict__ w,
                            const float* __restrict__ bias, float* __restrict__ y,
                            int Kw, int Lout) {
  int idx = blockIdx.x * blockDim.x + threadIdx.x;
  if (idx >= 4 * 64 * Lout) return;
  int l = idx % Lout;
  int oc = (idx / Lout) & 63;
  int b = idx / (Lout * 64);
  float acc = bias[oc];
  for (int t = 0; t < Kw; ++t) {
    const float* xr = x + (size_t)(b * 2048 + l + t) * 768;
    const float* wr = w + (size_t)oc * 768 * Kw + t;
    for (int c = 0; c < 768; ++c) acc += xr[c] * wr[c * Kw];
  }
  y[idx] = acc;  // layout [b][oc][l]
}

// ---------- batchnorm stats (training mode: over B,L) ----------
__global__ __launch_bounds__(256) void bn_stats_kernel(const float* __restrict__ y,
                                                       int Lout, float* __restrict__ mv) {
  __shared__ float sred[256];
  int ch = blockIdx.x;
  float s = 0.f, qq = 0.f;
  for (int t = threadIdx.x; t < 4 * Lout; t += 256) {
    int b = t / Lout, l = t % Lout;
    float vv = y[((size_t)b * 64 + ch) * Lout + l];
    s += vv; qq += vv * vv;
  }
  s = blk_sum(s, sred);
  qq = blk_sum(qq, sred);
  if (threadIdx.x == 0) {
    float n = (float)(4 * Lout);
    float mean = s / n;
    float var = qq / n - mean * mean;
    mv[ch * 2] = mean;
    mv[ch * 2 + 1] = rsqrtf(var + 1e-5f);
  }
}

// ---------- BN + ReLU + MaxPool(k=4,s=1) into cat buffer ----------
__global__ void bn_relu_pool_kernel(const float* __restrict__ y, const float* __restrict__ mv,
                                    const float* __restrict__ g, const float* __restrict__ bb,
                                    float* __restrict__ cat, int Lout, int Pout, int off, int T) {
  int idx = blockIdx.x * blockDim.x + threadIdx.x;
  if (idx >= 4 * 64 * Pout) return;
  int l = idx % Pout;
  int ch = (idx / Pout) & 63;
  int b = idx / (Pout * 64);
  float mean = mv[ch * 2], inv = mv[ch * 2 + 1];
  float gg = g[ch], bc = bb[ch];
  float m = -3.4e38f;
  for (int w = 0; w < 4; ++w) {
    float vv = (y[((size_t)b * 64 + ch) * Lout + l + w] - mean) * inv * gg + bc;
    vv = fmaxf(vv, 0.f);
    m = fmaxf(m, vv);
  }
  cat[((size_t)b * 64 + ch) * T + off + l] = m;
}

// ---------- pack xs[t,b,c] = cat[b,c,t] as bf16 for the LSTM input GEMM ----------
__global__ void xs_pack_kernel(const float* __restrict__ cat,
                               unsigned short* __restrict__ xsb, int T) {
  int idx = blockIdx.x * blockDim.x + threadIdx.x;
  if (idx >= T * 256) return;
  int t = idx >> 8;
  int rem = idx & 255;
  int b = rem >> 6, c = rem & 63;
  xsb[idx] = tobf16(cat[((size_t)b * 64 + c) * T + t]);
}

// ---------- bidirectional LSTM recurrence (xW precomputed) ----------
__global__ __launch_bounds__(256) void lstm_kernel(
    const float* __restrict__ xWf, const float* __restrict__ xWb,
    const float* __restrict__ whhf, const float* __restrict__ whhb,
    float* __restrict__ hf, float* __restrict__ hb, int T) {
  int dir = blockIdx.x;
  const float* xW = dir ? xWb : xWf;
  const float* Whh = dir ? whhb : whhf;
  float* hs = dir ? hb : hf;
  int j = threadIdx.x;
  float wrow[64];
  for (int kk = 0; kk < 64; ++kk) wrow[kk] = Whh[j * 64 + kk];
  __shared__ float hsm[4][64];
  __shared__ float z[4][256];
  hsm[j >> 6][j & 63] = 0.f;
  float c = 0.f;
  int bb = j >> 6, u = j & 63;
  __syncthreads();
  for (int step = 0; step < T; ++step) {
    int t = dir ? (T - 1 - step) : step;
    for (int b2 = 0; b2 < 4; ++b2) {
      float a = xW[((size_t)t * 4 + b2) * 256 + j];
      for (int kk = 0; kk < 64; ++kk) a += hsm[b2][kk] * wrow[kk];
      z[b2][j] = a;
    }
    __syncthreads();
    float zi = z[bb][u], zf = z[bb][64 + u], zg = z[bb][128 + u], zo = z[bb][192 + u];
    float fg = 1.f / (1.f + __expf(-zf));
    float ig = 1.f / (1.f + __expf(-zi));
    float og = 1.f / (1.f + __expf(-zo));
    c = fg * c + ig * tanhf(zg);
    float hv = og * tanhf(c);
    __syncthreads();
    hsm[bb][u] = hv;
    hs[((size_t)t * 4 + bb) * 64 + u] = hv;
    __syncthreads();
  }
}

// ---------- attention pooling: scores ----------
__global__ __launch_bounds__(128) void attnpool_scores_kernel(
    const float* __restrict__ hf, const float* __restrict__ hb,
    const float* __restrict__ aw1, const float* __restrict__ ab1,
    const float* __restrict__ aw2, const float* __restrict__ ab2,
    float* __restrict__ scores, int T) {
  int idx = blockIdx.x * blockDim.x + threadIdx.x;
  if (idx >= 4 * T) return;
  int b = idx / T, t = idx % T;
  float hcat[128];
  for (int i2 = 0; i2 < 64; ++i2) {
    hcat[i2] = hf[((size_t)t * 4 + b) * 64 + i2];
    hcat[64 + i2] = hb[((size_t)t * 4 + b) * 64 + i2];
  }
  float s = ab2[0];
  for (int jj = 0; jj < 64; ++jj) {
    float a = ab1[jj];
    for (int i2 = 0; i2 < 128; ++i2) a += hcat[i2] * aw1[i2 * 64 + jj];
    s += aw2[jj] * tanhf(a);
  }
  scores[idx] = s;
}

// ---------- attention pooling: softmax over T + weighted sum ----------
__global__ __launch_bounds__(128) void attnpool_out_kernel(
    const float* __restrict__ hf, const float* __restrict__ hb,
    const float* __restrict__ scores, float* __restrict__ out, int T) {
  __shared__ float sred[128];
  int b = blockIdx.x;
  int j = threadIdx.x;
  float m = -3.4e38f;
  for (int t = j; t < T; t += 128) m = fmaxf(m, scores[(size_t)b * T + t]);
  sred[j] = m; __syncthreads();
  for (int s2 = 64; s2 > 0; s2 >>= 1) {
    if (j < s2) sred[j] = fmaxf(sred[j], sred[j + s2]);
    __syncthreads();
  }
  m = sred[0]; __syncthreads();
  float zs = 0.f;
  for (int t = j; t < T; t += 128) zs += __expf(scores[(size_t)b * T + t] - m);
  sred[j] = zs; __syncthreads();
  for (int s2 = 64; s2 > 0; s2 >>= 1) {
    if (j < s2) sred[j] += sred[j + s2];
    __syncthreads();
  }
  float Z = sred[0]; __syncthreads();
  float accv = 0.f;
  const float* hsrc = (j < 64) ? hf : hb;
  int jj = j & 63;
  for (int t = 0; t < T; ++t) {
    float w = __expf(scores[(size_t)b * T + t] - m);
    accv += hsrc[((size_t)t * 4 + b) * 64 + jj] * w;
  }
  out[b * 128 + j] = accv / Z;
}

// ---------- host orchestration ----------
extern "C" void kernel_launch(void* const* d_in, const int* in_sizes, int n_in,
                              void* d_out, int out_size, void* d_ws, size_t ws_size,
                              hipStream_t stream) {
  (void)in_sizes; (void)n_in; (void)out_size; (void)ws_size;
  const int* ids = (const int*)d_in[0];
  const int* rnd = (const int*)d_in[1];
  auto F = [&](int i) { return (const float*)d_in[i]; };

  const int M8 = 8192, D = 768, FF = 3072;
  const int T = 6126;

  char* ws = (char*)d_ws;
  size_t off = 0;
  auto alloc = [&](size_t bytes) -> void* {
    void* p = ws + off;
    off += (bytes + 255) & ~(size_t)255;
    return p;
  };

  float* xf = (float*)alloc((size_t)M8 * D * 4);
  unsigned short* xb = (unsigned short*)alloc((size_t)M8 * D * 2);
  unsigned short* qb = (unsigned short*)alloc((size_t)M8 * D * 2);
  unsigned short* kb = (unsigned short*)alloc((size_t)M8 * D * 2);
  unsigned short* vb = (unsigned short*)alloc((size_t)M8 * D * 2);
  unsigned short* vT = (unsigned short*)alloc((size_t)M8 * D * 2);
  unsigned short* ob = (unsigned short*)alloc((size_t)M8 * D * 2);
  float* tf = (float*)alloc((size_t)M8 * D * 4);
  unsigned short* hb16 = (unsigned short*)alloc((size_t)M8 * FF * 2);
  unsigned short* wcvt = (unsigned short*)alloc((size_t)D * FF * 2);
  float* convy = (float*)alloc((size_t)4 * 64 * 2046 * 4);
  float* bnmv = (float*)alloc(64 * 2 * 4);
  float* cat = (float*)alloc((size_t)4 * 64 * T * 4);
  unsigned short* xsb = (unsigned short*)alloc((size_t)T * 256 * 2);
  unsigned short* wihTf = (unsigned short*)alloc(64 * 256 * 2);
  unsigned short* wihTb = (unsigned short*)alloc(64 * 256 * 2);
  float* xWf = (float*)alloc((size_t)T * 4 * 256 * 4);
  float* xWb = (float*)alloc((size_t)T * 4 * 256 * 4);
  float* hfb = (float*)alloc((size_t)T * 4 * 64 * 4);
  float* hbb = (float*)alloc((size_t)T * 4 * 64 * 4);
  float* scores = (float*)alloc((size_t)4 * T * 4);

  auto wtrans = [&](const float* s, unsigned short* d, int K, int N) {
    transpose_f2bf_kernel<<<(K * N + 255) / 256, 256, 0, stream>>>(s, d, K, N);
  };
  auto gemm = [&](const unsigned short* A, const unsigned short* WT,
                  const float* bias, const float* bias2, float* Cf,
                  unsigned short* Cb, int M, int N, int K, int act) {
    dim3 g(N / 128, (M + 127) / 128);
    if (act == 1)
      gemm_bf16_kernel<1, false, false, true><<<g, 256, 0, stream>>>(
          A, WT, bias, nullptr, nullptr, Cb, M, N, K);
    else if (bias2)
      gemm_bf16_kernel<0, true, true, false><<<g, 256, 0, stream>>>(
          A, WT, bias, bias2, Cf, nullptr, M, N, K);
    else if (Cb)
      gemm_bf16_kernel<0, false, false, true><<<g, 256, 0, stream>>>(
          A, WT, bias, nullptr, nullptr, Cb, M, N, K);
    else
      gemm_bf16_kernel<0, false, true, false><<<g, 256, 0, stream>>>(
          A, WT, bias, nullptr, Cf, nullptr, M, N, K);
  };

  // embedding + LN
  embed_ln_kernel<<<M8, 256, 0, stream>>>(ids, F(2), F(3), F(4), F(5), xf, xb);

  // 2 BigBird layers
  for (int L = 0; L < 2; ++L) {
    int base = 6 + L * 16;
    const float *wq = F(base + 0), *bq = F(base + 1), *wk = F(base + 2), *bk = F(base + 3),
                *wv = F(base + 4), *bv = F(base + 5), *wo = F(base + 6), *bo = F(base + 7),
                *l1g = F(base + 8), *l1b = F(base + 9), *w1 = F(base + 10), *b1 = F(base + 11),
                *w2 = F(base + 12), *b2 = F(base + 13), *l2g = F(base + 14), *l2b = F(base + 15);

    wtrans(wq, wcvt, D, D);
    gemm(xb, wcvt, bq, nullptr, nullptr, qb, M8, D, D, 0);
    wtrans(wk, wcvt, D, D);
    gemm(xb, wcvt, bk, nullptr, nullptr, kb, M8, D, D, 0);
    wtrans(wv, wcvt, D, D);
    gemm(xb, wcvt, bv, nullptr, nullptr, vb, M8, D, D, 0);
    vtrans_kernel<<<(M8 * D + 255) / 256, 256, 0, stream>>>(vb, vT);

    attn_kernel<<<dim3(32, 12, 4), 128, 0, stream>>>(qb, kb, vT, rnd, ob);

    wtrans(wo, wcvt, D, D);
    gemm(ob, wcvt, bo, nullptr, tf, nullptr, M8, D, D, 0);
    add_ln_kernel<<<M8, 256, 0, stream>>>(xf, tf, l1g, l1b, xb);

    wtrans(w1, wcvt, D, FF);
    gemm(xb, wcvt, b1, nullptr, nullptr, hb16, M8, FF, D, 1);  // GELU
    wtrans(w2, wcvt, FF, D);
    gemm(hb16, wcvt, b2, nullptr, tf, nullptr, M8, D, FF, 0);
    add_ln_kernel<<<M8, 256, 0, stream>>>(xf, tf, l2g, l2b, xb);
  }

  // conv branches -> BN(train) -> ReLU -> MaxPool -> cat
  const int Louts[3] = {2046, 2045, 2044};
  const int Kws[3] = {3, 4, 5};
  const int Pouts[3] = {2043, 2042, 2041};
  const int offs[3] = {0, 2043, 4085};
  const int widx[3] = {38, 40, 42}, bidx[3] = {39, 41, 43};
  const int bng[3] = {44, 46, 48}, bnb[3] = {45, 47, 49};
  for (int br = 0; br < 3; ++br) {
    int Lout = Louts[br], Pout = Pouts[br];
    int tot = 4 * 64 * Lout;
    conv_kernel<<<(tot + 255) / 256, 256, 0, stream>>>(xf, F(widx[br]), F(bidx[br]),
                                                       convy, Kws[br], Lout);
    bn_stats_kernel<<<64, 256, 0, stream>>>(convy, Lout, bnmv);
    int tot2 = 4 * 64 * Pout;
    bn_relu_pool_kernel<<<(tot2 + 255) / 256, 256, 0, stream>>>(
        convy, bnmv, F(bng[br]), F(bnb[br]), cat, Lout, Pout, offs[br], T);
  }

  // LSTM: precompute xW via WMMA GEMM (Wih is already [N=256][K=64] = W^T), then scan
  xs_pack_kernel<<<(T * 256 + 255) / 256, 256, 0, stream>>>(cat, xsb, T);
  f2bf_kernel<<<(256 * 64 + 255) / 256, 256, 0, stream>>>(F(50), wihTf, 256 * 64);
  f2bf_kernel<<<(256 * 64 + 255) / 256, 256, 0, stream>>>(F(54), wihTb, 256 * 64);
  gemm(xsb, wihTf, F(52), F(53), xWf, nullptr, T * 4, 256, 64, 0);
  gemm(xsb, wihTb, F(56), F(57), xWb, nullptr, T * 4, 256, 64, 0);
  lstm_kernel<<<2, 256, 0, stream>>>(xWf, xWb, F(51), F(55), hfb, hbb, T);

  // attention pooling
  attnpool_scores_kernel<<<(4 * T + 127) / 128, 128, 0, stream>>>(
      hfb, hbb, F(58), F(59), F(60), F(61), scores, T);
  attnpool_out_kernel<<<4, 128, 0, stream>>>(hfb, hbb, scores, (float*)d_out, T);
}
